// BatchedGNNModel_58523224375656
// MI455X (gfx1250) — matrix operand order repaired and checked by reading, not compiled, not run on hardware
//
#include <hip/hip_runtime.h>

// ---------------------------------------------------------------------------
// BatchedGNNModel on MI455X (gfx1250, wave32, WMMA + async global->LDS)
//   4x [ Y = H @ W^T + b  (WMMA f16 -> f32 acc, W tile staged in LDS) ;
//        H' = act(A_norm_sparse @ Y) ]
// ---------------------------------------------------------------------------

#define BATCH  64
#define NVERT  256
#define NNODE  768           // 3 * NVERT
#define BN     (BATCH * NNODE)
#define MAXD   8             // max row degree of A (actual max ~5)

#define LDS_ROW_PAD 16                       // bytes; avoids bank-group aliasing
#define LDS_ROW_MAX (256 * 2 + LDS_ROW_PAD)  // Kp<=256

typedef __attribute__((ext_vector_type(16))) _Float16 v16h;
typedef __attribute__((ext_vector_type(8)))  _Float16 v8h;
typedef __attribute__((ext_vector_type(8)))  float    v8f;

// GCC-style int4 pointee matching the async-builtin's parameter type:
//   param0: int4 __device__* (AS1 global), param1: int4 __shared__* (AS3 LDS)
typedef int v4i_vs __attribute__((vector_size(16)));
typedef __attribute__((address_space(1))) v4i_vs* gptr_b128;
typedef __attribute__((address_space(3))) v4i_vs* lptr_b128;

// ---------------------------------------------------------------------------
// 1) Scan adjacency rows: degree, d = deg^-1/2, and per-row nonzero lists.
// ---------------------------------------------------------------------------
__global__ void build_csr(const float* __restrict__ A,
                          int* __restrict__ nidx, float* __restrict__ nval,
                          int* __restrict__ ncnt, float* __restrict__ dvals)
{
    int i = blockIdx.x * blockDim.x + threadIdx.x;
    if (i >= NNODE) return;
    float deg = 0.0f;
    int cnt = 0;
    for (int j = 0; j < NNODE; ++j) {
        float a = A[(size_t)i * NNODE + j];
        if (a != 0.0f) {
            deg += a;
            if (cnt < MAXD) {
                nidx[i * MAXD + cnt] = j;
                nval[i * MAXD + cnt] = a;
                ++cnt;
            }
        }
    }
    ncnt[i]  = cnt;
    dvals[i] = (deg == 0.0f) ? 0.0f : rsqrtf(deg);
}

// ---------------------------------------------------------------------------
// 2) Normalized edge weights  w[i,k] = A[i,j] * d[i] * d[j]
// ---------------------------------------------------------------------------
__global__ void build_w(const int* __restrict__ nidx, const int* __restrict__ ncnt,
                        const float* __restrict__ nval, const float* __restrict__ dvals,
                        float* __restrict__ nw)
{
    int t = blockIdx.x * blockDim.x + threadIdx.x;
    if (t >= NNODE * MAXD) return;
    int i = t / MAXD, k = t % MAXD;
    float w = 0.0f;
    if (k < ncnt[i]) {
        int j = nidx[t];
        w = nval[t] * dvals[i] * dvals[j];
    }
    nw[t] = w;
}

// ---------------------------------------------------------------------------
// 3) Convert + zero-pad weights to f16, row-major (Npad x Kpad); row n holds
//    output column n's K-vector contiguously (this IS the WMMA B-fragment run).
// ---------------------------------------------------------------------------
__global__ void conv_w(const float* __restrict__ Wsrc, int No, int Ki,
                       _Float16* __restrict__ Wdst, int Npad, int Kpad)
{
    int t = blockIdx.x * blockDim.x + threadIdx.x;
    if (t >= Npad * Kpad) return;
    int n = t / Kpad, k = t % Kpad;
    _Float16 v = (_Float16)0.0f;
    if (n < No && k < Ki) v = (_Float16)Wsrc[(size_t)n * Ki + k];
    Wdst[t] = v;
}

__global__ void pad_bias(const float* __restrict__ bsrc, int n,
                         float* __restrict__ bdst, int npad)
{
    int t = blockIdx.x * blockDim.x + threadIdx.x;
    if (t >= npad) return;
    bdst[t] = (t < n) ? bsrc[t] : 0.0f;
}

// ---------------------------------------------------------------------------
// 4) Clamp grasped vertices into the first 3 features, convert to f16,
//    zero-pad features 6..31 so layer-1 is a single K=32 WMMA step.
// ---------------------------------------------------------------------------
__global__ void prep_h0(const float* __restrict__ x, const float* __restrict__ inp,
                        const int* __restrict__ psel, int np,
                        const int* __restrict__ c1sel, int nc1,
                        const int* __restrict__ c2sel, int nc2,
                        _Float16* __restrict__ H0)     // (B*N) x 32
{
    int bi = blockIdx.x * blockDim.x + threadIdx.x;
    if (bi >= BN) return;
    int i = bi % NNODE;
    bool clampv = false;
    for (int k = 0; k < np;  ++k) if (i ==            psel[k])  clampv = true;
    for (int k = 0; k < nc1; ++k) if (i ==  NVERT + c1sel[k])   clampv = true;
    for (int k = 0; k < nc2; ++k) if (i == 2*NVERT + c2sel[k])  clampv = true;
    const float* xr = x   + (size_t)bi * 6;
    const float* ir = inp + (size_t)bi * 3;
    _Float16* o = H0 + (size_t)bi * 32;
    o[0] = (_Float16)(clampv ? ir[0] : xr[0]);
    o[1] = (_Float16)(clampv ? ir[1] : xr[1]);
    o[2] = (_Float16)(clampv ? ir[2] : xr[2]);
    o[3] = (_Float16)xr[3];
    o[4] = (_Float16)xr[4];
    o[5] = (_Float16)xr[5];
    #pragma unroll
    for (int k = 6; k < 32; ++k) o[k] = (_Float16)0.0f;
}

// ---------------------------------------------------------------------------
// 5) WMMA GEMM:  Y(M x Np) = Hin(M x Kp) @ Wt(Np x Kp)^T + bias
//
//    Grid: (M/128, Np/16). Block = 8 waves; wave w -> rows [m+16w, m+16w+16).
//    The block's 16 x Kp weight tile (one contiguous 32*Kp-byte run of Wt) is
//    staged into LDS once via GLOBAL_LOAD_ASYNC_TO_LDS_B128 (ASYNCcnt ->
//    s_wait_asynccnt -> barrier) and reused by all 8 waves, with a 16-byte
//    per-row pad so the 16 B-fragment columns land on distinct bank groups.
//
//    Fragment layouts per CDNA5 ISA 7.12.2 (wave32):
//      A: lane -> row (lane&15); halves = K[8*(lane>>4) .. +8) ++ K[+16 .. +24)
//      B: lane -> col (lane&15); halves = row[col] K[8*(lane>>4) .. +8) ++ K[+16 .. +24)
//      C: vgpr r -> row r + 8*(lane>>4), col lane&15
// ---------------------------------------------------------------------------
__global__ __launch_bounds__(256)
void gemm_bias_f16(const _Float16* __restrict__ Hin,
                   const _Float16* __restrict__ Wt,
                   const float*    __restrict__ bias,
                   _Float16* __restrict__ Yout,
                   int M, int Kp, int Np)
{
    __shared__ __align__(64) char wtile[16 * LDS_ROW_MAX];

    const int nbase   = blockIdx.y << 4;
    const int mblock  = blockIdx.x << 7;          // 128 rows per block
    const int lane    = threadIdx.x & 31;
    const int wave    = threadIdx.x >> 5;

    const int rowBytes     = Kp << 1;             // bytes per weight row
    const int ldsRowBytes  = rowBytes + LDS_ROW_PAD;
    const int chunksPerRow = rowBytes >> 4;       // 16B chunks per row
    const int totalChunks  = chunksPerRow << 4;   // x 16 rows (contig in global)

    // ---- stage 16 x Kp weight tile into LDS (async DMA, ASYNCcnt) ----
    {
        const char* gsrc = (const char*)(Wt + (size_t)nbase * Kp);
        for (int c = (int)threadIdx.x; c < totalChunks; c += 256) {
            const int row  = c / chunksPerRow;
            const int col  = c - row * chunksPerRow;
            const int goff = c << 4;                            // contiguous
            const int loff = row * ldsRowBytes + (col << 4);
#if __has_builtin(__builtin_amdgcn_global_load_async_to_lds_b128)
            __builtin_amdgcn_global_load_async_to_lds_b128(
                (gptr_b128)(uintptr_t)(gsrc + goff),
                (lptr_b128)(unsigned)(uintptr_t)(wtile + loff),
                0, 0);
#else
            *(v8h*)(wtile + loff) = *(const v8h*)(gsrc + goff);
#endif
        }
#if __has_builtin(__builtin_amdgcn_s_wait_asynccnt)
        __builtin_amdgcn_s_wait_asynccnt(0);
#endif
        __syncthreads();
    }

    const int rc = lane & 15;     // A-row / B-col / C-col for this lane
    const int hi = lane >> 4;     // K-half select
    const int mbase = mblock + (wave << 4);

    const _Float16* ap  = Hin + (size_t)(mbase + rc) * Kp + hi * 8;
    const char*     lbp = wtile + rc * ldsRowBytes + (hi << 4);

    v8f acc = {};
    for (int k0 = 0; k0 < Kp; k0 += 32) {
        __builtin_prefetch((const void*)(ap + k0 + 32), 0, 0);
        union { v16h v; v8h h[2]; } au, bu;
        au.h[0] = *(const v8h*)(ap + k0);          // K = hi*8      .. +8
        au.h[1] = *(const v8h*)(ap + k0 + 16);     // K = hi*8 + 16 .. +8
        bu.h[0] = *(const v8h*)(lbp + (k0 << 1));  // K = hi*8      .. +8  (LDS)
        bu.h[1] = *(const v8h*)(lbp + (k0 << 1) + 32);
        acc = __builtin_amdgcn_wmma_f32_16x16x32_f16(
                  false, au.v, false, bu.v, (short)0, acc, false, false);
    }

    const float bv = bias[nbase + rc];
    _Float16* outp = Yout + (size_t)(mbase + hi * 8) * Np + (nbase + rc);
    #pragma unroll
    for (int r = 0; r < 8; ++r)
        outp[(size_t)r * Np] = (_Float16)(acc[r] + bv);
}

// ---------------------------------------------------------------------------
// 6) Sparse normalized aggregation:  H'[b,i,f] = act( sum_k w[i,k]*Y[b,j_k,f] )
//    One block per (b,i), one thread per feature.  Degree <= 8 gather.
// ---------------------------------------------------------------------------
__global__ void agg_relu_f16(const _Float16* __restrict__ Yin,
                             const int* __restrict__ nidx, const float* __restrict__ nw,
                             const int* __restrict__ ncnt,
                             _Float16* __restrict__ Hout, int F, int doRelu)
{
    const int f  = threadIdx.x;
    const int bi = blockIdx.x;           // b*NNODE + i
    if (f >= F) return;
    const int i = bi % NNODE;
    const int b = bi / NNODE;
    float accv = 0.0f;
    const int cnt = ncnt[i];
    for (int k = 0; k < cnt; ++k) {
        const int   j = nidx[i * MAXD + k];
        const float w = nw[i * MAXD + k];
        accv += w * (float)Yin[((size_t)b * NNODE + j) * F + f];
    }
    if (doRelu) accv = fmaxf(accv, 0.0f);
    Hout[(size_t)bi * F + f] = (_Float16)accv;
}

// ---------------------------------------------------------------------------
// 7) Final aggregation to f32 output (B,N,3); Y4 is padded to 16 columns.
// ---------------------------------------------------------------------------
__global__ void agg_out_f32(const _Float16* __restrict__ Y4,
                            const int* __restrict__ nidx, const float* __restrict__ nw,
                            const int* __restrict__ ncnt, float* __restrict__ out)
{
    int t = blockIdx.x * blockDim.x + threadIdx.x;
    if (t >= BN * 3) return;
    int c  = t % 3;
    int bi = t / 3;
    int i  = bi % NNODE;
    int b  = bi / NNODE;
    float accv = 0.0f;
    const int cnt = ncnt[i];
    for (int k = 0; k < cnt; ++k) {
        int j = nidx[i * MAXD + k];
        accv += nw[i * MAXD + k] * (float)Y4[((size_t)b * NNODE + j) * 16 + c];
    }
    out[t] = accv;
}

// ---------------------------------------------------------------------------
// 8) Overwrite grasped vertices of the output with `inputs` (exact f32 copy).
// ---------------------------------------------------------------------------
__global__ void clamp_out(const float* __restrict__ inp,
                          const int* __restrict__ psel, int np,
                          const int* __restrict__ c1sel, int nc1,
                          const int* __restrict__ c2sel, int nc2,
                          float* __restrict__ out)
{
    int b = blockIdx.x;
    int t = threadIdx.x;
    int total = (np + nc1 + nc2) * 3;
    if (t >= total) return;
    int s = t / 3, c = t % 3;
    int n;
    if (s < np)            n =             psel[s];
    else if (s < np + nc1) n =  NVERT  + c1sel[s - np];
    else                   n = 2*NVERT + c2sel[s - np - nc1];
    size_t off = ((size_t)b * NNODE + n) * 3 + c;
    out[off] = inp[off];
}

// ---------------------------------------------------------------------------
// Host-side orchestration (graph-capture safe: only kernel launches on stream)
// ---------------------------------------------------------------------------
extern "C" void kernel_launch(void* const* d_in, const int* in_sizes, int n_in,
                              void* d_out, int out_size, void* d_ws, size_t ws_size,
                              hipStream_t stream)
{
    (void)n_in; (void)out_size; (void)ws_size;
    const float* x    = (const float*)d_in[0];
    const float* inp  = (const float*)d_in[1];
    const float* A    = (const float*)d_in[2];
    const float* W1   = (const float*)d_in[3];
    const float* b1   = (const float*)d_in[4];
    const float* W2   = (const float*)d_in[5];
    const float* b2   = (const float*)d_in[6];
    const float* W3   = (const float*)d_in[7];
    const float* b3   = (const float*)d_in[8];
    const float* W4   = (const float*)d_in[9];
    const float* b4   = (const float*)d_in[10];
    const int* psel   = (const int*)d_in[11];
    const int* c1sel  = (const int*)d_in[12];
    const int* c2sel  = (const int*)d_in[13];
    const int np  = in_sizes[11];
    const int nc1 = in_sizes[12];
    const int nc2 = in_sizes[13];
    float* out = (float*)d_out;

    // workspace carve-out (256B aligned slices)
    char* wsb = (char*)d_ws;
    size_t off = 0;
    auto carve = [&](size_t bytes) -> char* {
        char* p = wsb + off;
        off = (off + bytes + 255) & ~(size_t)255;
        return p;
    };
    float*    dvals = (float*)   carve(NNODE * 4);
    int*      nidx  = (int*)     carve(NNODE * MAXD * 4);
    float*    nval  = (float*)   carve(NNODE * MAXD * 4);
    float*    nw    = (float*)   carve(NNODE * MAXD * 4);
    int*      ncnt  = (int*)     carve(NNODE * 4);
    _Float16* Wp1   = (_Float16*)carve((size_t)256 * 32  * 2);
    _Float16* Wp2   = (_Float16*)carve((size_t)128 * 256 * 2);
    _Float16* Wp3   = (_Float16*)carve((size_t)128 * 128 * 2);
    _Float16* Wp4   = (_Float16*)carve((size_t)16  * 128 * 2);
    float*    bp4   = (float*)   carve(16 * 4);
    _Float16* bufA  = (_Float16*)carve((size_t)BN * 256 * 2);   // ping
    _Float16* bufB  = (_Float16*)carve((size_t)BN * 256 * 2);   // pong
    _Float16* bufC  = (_Float16*)carve((size_t)BN * 256 * 2);   // pang

    // --- graph prep ---
    build_csr<<<(NNODE + 255) / 256, 256, 0, stream>>>(A, nidx, nval, ncnt, dvals);
    build_w<<<(NNODE * MAXD + 255) / 256, 256, 0, stream>>>(nidx, ncnt, nval, dvals, nw);

    // --- weight prep (f16, padded, B-fragment friendly) ---
    conv_w<<<(256 * 32  + 255) / 256, 256, 0, stream>>>(W1, 256, 6,   Wp1, 256, 32);
    conv_w<<<(128 * 256 + 255) / 256, 256, 0, stream>>>(W2, 128, 256, Wp2, 128, 256);
    conv_w<<<(128 * 128 + 255) / 256, 256, 0, stream>>>(W3, 128, 128, Wp3, 128, 128);
    conv_w<<<(16  * 128 + 255) / 256, 256, 0, stream>>>(W4, 3,   128, Wp4, 16,  128);
    pad_bias<<<1, 16, 0, stream>>>(b4, 3, bp4, 16);

    // --- inputs (clamped, f16, K padded to 32) ---
    prep_h0<<<(BN + 255) / 256, 256, 0, stream>>>(x, inp, psel, np, c1sel, nc1,
                                                  c2sel, nc2, bufA);

    // --- layer 1: 6(->32) -> 256, ReLU ---
    gemm_bias_f16<<<dim3(BN / 128, 256 / 16), 256, 0, stream>>>(bufA, Wp1, b1, bufC,
                                                                BN, 32, 256);
    agg_relu_f16<<<BN, 256, 0, stream>>>(bufC, nidx, nw, ncnt, bufB, 256, 1);

    // --- layer 2: 256 -> 128, ReLU ---
    gemm_bias_f16<<<dim3(BN / 128, 128 / 16), 256, 0, stream>>>(bufB, Wp2, b2, bufA,
                                                                BN, 256, 128);
    agg_relu_f16<<<BN, 128, 0, stream>>>(bufA, nidx, nw, ncnt, bufC, 128, 1);

    // --- layer 3: 128 -> 128, no activation ---
    gemm_bias_f16<<<dim3(BN / 128, 128 / 16), 256, 0, stream>>>(bufC, Wp3, b3, bufA,
                                                                BN, 128, 128);
    agg_relu_f16<<<BN, 128, 0, stream>>>(bufA, nidx, nw, ncnt, bufB, 128, 0);

    // --- layer 4: 128 -> 3(->16), aggregate straight to f32 output ---
    gemm_bias_f16<<<dim3(BN / 128, 1), 256, 0, stream>>>(bufB, Wp4, bp4, bufA,
                                                         BN, 128, 16);
    agg_out_f32<<<(BN * 3 + 255) / 256, 256, 0, stream>>>(bufA, nidx, nw, ncnt, out);

    // --- clamp grasped vertices in the output ---
    clamp_out<<<BATCH, 32, 0, stream>>>(inp, psel, np, c1sel, nc1, c2sel, nc2, out);
}
